// SparseLayer_17789754540032
// MI455X (gfx1250) — matrix operand order repaired and checked by reading, not compiled, not run on hardware
//
#include <hip/hip_runtime.h>
#include <stdint.h>

#define IN_F   8192
#define OUT_F  8192
#define BATCH  32

typedef _Float16 v16h __attribute__((ext_vector_type(16)));
typedef _Float16 v8h  __attribute__((ext_vector_type(8)));
typedef float    v8f  __attribute__((ext_vector_type(8)));

__device__ __forceinline__ void atomicAddF32(float* p, float v) {
    __hip_atomic_fetch_add(p, v, __ATOMIC_RELAXED, __HIP_MEMORY_SCOPE_AGENT);
}

// Packed-f16 atomic add (CDNA5 GLOBAL_ATOMIC_PK_ADD_F16) into element idx of an
// f16 array. The 4B-aligned pair base is idx&~1; value goes in lo/hi by parity.
__device__ __forceinline__ void pkAddF16(_Float16* base, uint32_t idx, float v) {
    uint16_t hb   = __builtin_bit_cast(uint16_t, (_Float16)v);
    uint32_t data = (uint32_t)hb << ((idx & 1u) << 4);
    uint64_t addr = (uint64_t)(uintptr_t)(base + (idx & ~1u));
    asm volatile("global_atomic_pk_add_f16 %0, %1, off scope:SCOPE_DEV"
                 :: "v"(addr), "v"(data) : "memory");
}

// ---------- Kernel 1: zero the f16 weight scratch (128 MB, L2-resident) ----------
__global__ void __launch_bounds__(256) k_zero_w(uint4* __restrict__ W16) {
    const size_t n = (size_t)OUT_F * IN_F * 2 / 16;   // bytes/16
    uint4 z = {0u, 0u, 0u, 0u};
    for (size_t j = (size_t)blockIdx.x * blockDim.x + threadIdx.x; j < n;
         j += (size_t)gridDim.x * blockDim.x)
        W16[j] = z;
}

// ---------- Kernel 2: out = bias, and pre-convert in_values -> f16 ----------
__global__ void __launch_bounds__(256) k_init(float* __restrict__ out,
                                              const float* __restrict__ bias,
                                              const float* __restrict__ in,
                                              _Float16* __restrict__ inH) {
    int i = blockIdx.x * blockDim.x + threadIdx.x;
    if (i < BATCH * OUT_F) {
        out[i] = bias[i & (OUT_F - 1)];
        inH[i] = (_Float16)in[i];
    }
}

// ---------- Kernel 3: COO scatter-densify: Wh[r,c] += v (pk f16 atomics) ----------
__global__ void __launch_bounds__(256) k_scatter(const float* __restrict__ vals,
                                                 const int*   __restrict__ rows,
                                                 const int*   __restrict__ cols,
                                                 _Float16* __restrict__ Wh, int nnz) {
    int i = 4 * (blockIdx.x * blockDim.x + threadIdx.x);
    if (i >= nnz) return;
    if (i + 4 <= nnz) {
        float4 v = *(const float4*)(vals + i);
        int4   r = *(const int4*)(rows + i);
        int4   c = *(const int4*)(cols + i);
        pkAddF16(Wh, (uint32_t)r.x * IN_F + (uint32_t)c.x, v.x);
        pkAddF16(Wh, (uint32_t)r.y * IN_F + (uint32_t)c.y, v.y);
        pkAddF16(Wh, (uint32_t)r.z * IN_F + (uint32_t)c.z, v.z);
        pkAddF16(Wh, (uint32_t)r.w * IN_F + (uint32_t)c.w, v.w);
    } else {
        for (int t = i; t < nnz; ++t)
            pkAddF16(Wh, (uint32_t)rows[t] * IN_F + (uint32_t)cols[t], vals[t]);
    }
}

// ---------- Kernel 4: WMMA GEMM  out[b,n] += sum_k inH[b,k]*Wh[n,k] ----------
// Wave: one 16-wide N-tile, full M=32 (two 16x16 D tiles sharing B), K-chunk 1024.
// A layout (16x32 f16): lane l row m=l&15; e<8 -> k+8*khi+e ; e>=8 -> k+16+8*khi+(e-8)
// B layout (32x16): lane l holds col n=l&15 = W row n, same k order as A
__global__ void __launch_bounds__(128) k_wmma_gemm(const _Float16* __restrict__ A,  // [32][8192] f16
                                                   const _Float16* __restrict__ W,  // [8192][8192] f16
                                                   float* __restrict__ out) {       // [32][8192] f32
    const int lane  = threadIdx.x & 31;
    const int wave  = threadIdx.x >> 5;
    const int ntile = blockIdx.x * 4 + wave;            // 0..511
    const int kbase = blockIdx.y * (IN_F / 8);          // chunks of 1024
    const int khi   = lane >> 4;
    const int nl    = lane & 15;
    const int ncol  = ntile * 16 + nl;

    const _Float16* arow0 = A + (size_t)nl * IN_F;          // M = nl
    const _Float16* arow1 = A + (size_t)(nl + 16) * IN_F;   // M = nl + 16
    const _Float16* wrow  = W + (size_t)ncol * IN_F;        // B[k][n] = W[n][k]

    v8f c0 = {}; v8f c1 = {};

    for (int k = kbase; k < kbase + IN_F / 8; k += 32) {
        v8h a0l = *(const v8h*)(arow0 + k + 8 * khi);
        v8h a0h = *(const v8h*)(arow0 + k + 16 + 8 * khi);
        v8h a1l = *(const v8h*)(arow1 + k + 8 * khi);
        v8h a1h = *(const v8h*)(arow1 + k + 16 + 8 * khi);
        v8h bl = *(const v8h*)(wrow + k + 8 * khi);
        v8h bh = *(const v8h*)(wrow + k + 16 + 8 * khi);
        v16h vb = __builtin_shufflevector(bl, bh, 0,1,2,3,4,5,6,7,8,9,10,11,12,13,14,15);

        v16h va0 = __builtin_shufflevector(a0l, a0h, 0,1,2,3,4,5,6,7,8,9,10,11,12,13,14,15);
        v16h va1 = __builtin_shufflevector(a1l, a1h, 0,1,2,3,4,5,6,7,8,9,10,11,12,13,14,15);

        // stream-ahead prefetch of this lane's W row (global_prefetch_b8)
        __builtin_prefetch(wrow + k + 1024, 0, 1);

        c0 = __builtin_amdgcn_wmma_f32_16x16x32_f16(false, va0, false, vb,
                                                    (short)0, c0, false, false);
        c1 = __builtin_amdgcn_wmma_f32_16x16x32_f16(false, va1, false, vb,
                                                    (short)0, c1, false, false);
    }

    // D layout: VGPR j -> M = j + 8*khi, column = ncol
#pragma unroll
    for (int j = 0; j < 8; ++j) {
        atomicAddF32(out + (size_t)(j + 8 * khi) * OUT_F + ncol,      c0[j]);
        atomicAddF32(out + (size_t)(16 + j + 8 * khi) * OUT_F + ncol, c1[j]);
    }
}

extern "C" void kernel_launch(void* const* d_in, const int* in_sizes, int n_in,
                              void* d_out, int out_size, void* d_ws, size_t ws_size,
                              hipStream_t stream) {
    const float* in_values = (const float*)d_in[0];   // [32,8192] f32
    const float* values    = (const float*)d_in[1];   // [NNZ] f32
    const int*   indices   = (const int*)d_in[2];     // [2,NNZ] i32
    const float* bias      = (const float*)d_in[3];   // [8192] f32
    float*       out       = (float*)d_out;           // [32,8192] f32

    // Workspace layout: Wh f16 [8192*8192] = 128 MB, then inH f16 [32*8192] = 512 KB
    _Float16* Wh  = (_Float16*)d_ws;
    _Float16* inH = Wh + (size_t)OUT_F * IN_F;

    const int  nnz  = in_sizes[1];
    const int* rows = indices;
    const int* cols = indices + nnz;

    k_zero_w<<<dim3(2048), dim3(256), 0, stream>>>((uint4*)Wh);
    k_init<<<dim3((BATCH * OUT_F + 255) / 256), dim3(256), 0, stream>>>(out, bias, in_values, inH);

    int sthreads = (nnz + 3) / 4;
    k_scatter<<<dim3((sthreads + 255) / 256), dim3(256), 0, stream>>>(values, rows, cols, Wh, nnz);

    k_wmma_gemm<<<dim3(OUT_F / 16 / 4, 8), dim3(128), 0, stream>>>(inH, Wh, out);
}
